// Attention_32246614458829
// MI455X (gfx1250) — compile-verified, matrix operand order checked by reading, not run on hardware
//
#include <hip/hip_runtime.h>
#include <hip/hip_bf16.h>

#define DIM    256
#define NHEADS 8
#define HD     32
#define KLEN   512
#define NTILES (KLEN / 16)
#define EPSF   1e-5f

typedef __attribute__((ext_vector_type(16))) _Float16 v16h;
typedef __attribute__((ext_vector_type(8)))  float    v8f;
typedef __attribute__((ext_vector_type(4)))  int      v4i;

#ifndef __has_builtin
#define __has_builtin(x) 0
#endif
#if __has_builtin(__builtin_amdgcn_global_load_async_to_lds_b128) && \
    __has_builtin(__builtin_amdgcn_s_wait_asynccnt)
#define HAVE_ASYNC_LDS 1
#else
#define HAVE_ASYNC_LDS 0
#endif

#define AS1 __attribute__((address_space(1)))
#define AS3 __attribute__((address_space(3)))

// ---- async staging helpers (ASYNCcnt-tracked copy engine, CDNA5) ----------
__device__ __forceinline__ void wait_async4() {
#if HAVE_ASYNC_LDS
    __builtin_amdgcn_s_wait_asynccnt(4);   // one 4-instruction batch in flight
#endif
}
__device__ __forceinline__ void wait_async0() {
#if HAVE_ASYNC_LDS
    __builtin_amdgcn_s_wait_asynccnt(0);   // drain
#endif
}

// Stage one 16x256 fp32 tile (16 KB) into LDS: 256 threads x 4 x b128.
__device__ __forceinline__ void stage_raw(const float* __restrict__ src_base,
                                          float* raw, int t) {
    const int srow = t >> 4, scol = (t & 15) * 16;
    const float* src = src_base + srow * DIM + scol;
    float*       dst = raw      + srow * DIM + scol;
#if HAVE_ASYNC_LDS
#pragma unroll
    for (int j = 0; j < 4; ++j)
        __builtin_amdgcn_global_load_async_to_lds_b128(
            (AS1 v4i*)(src + j * 4), (AS3 v4i*)(dst + j * 4), 0, 0);
#else
#pragma unroll
    for (int j = 0; j < 4; ++j)
        ((float4*)dst)[j] = ((const float4*)src)[j];
#endif
}

// Build a WMMA f16 operand (A or B) per the CDNA5 16-bit operand layout:
//   lane L (L<16):  row = L,    K = {base..base+7, base+16..base+23}
//   lane L (L>=16): row = L-16, K pattern shifted by +8
// `p` must already point at row_base + kbase + (hi ? 8 : 0).
__device__ __forceinline__ v16h make_op16(const _Float16* p) {
    v16h a;
#pragma unroll
    for (int j = 0; j < 8; ++j) a[j] = p[j];          // 16B chunk
#pragma unroll
    for (int j = 0; j < 8; ++j) a[8 + j] = p[16 + j]; // 16B chunk
    return a;
}

__device__ __forceinline__ v8f wmma16(v16h a, v16h b, v8f c) {
    return __builtin_amdgcn_wmma_f32_16x16x32_f16(
        false, a, false, b, (short)0, c, false, false);
}

// ---------------------------------------------------------------- kernel 1
__global__ void cvt_weights(const float* __restrict__ Wk,
                            const float* __restrict__ Wv,
                            _Float16* __restrict__ wk16,
                            _Float16* __restrict__ wv16) {
    int i = (blockIdx.x * blockDim.x + threadIdx.x) * 4;
    if (i < DIM * DIM) {
        float4 a = *(const float4*)(Wk + i);
        wk16[i + 0] = (_Float16)a.x; wk16[i + 1] = (_Float16)a.y;
        wk16[i + 2] = (_Float16)a.z; wk16[i + 3] = (_Float16)a.w;
        float4 b = *(const float4*)(Wv + i);
        wv16[i + 0] = (_Float16)b.x; wv16[i + 1] = (_Float16)b.y;
        wv16[i + 2] = (_Float16)b.z; wv16[i + 3] = (_Float16)b.w;
    }
}

// ---------------------------------------------------------------- kernel 2
// q = LN_head(x_q @ Wq^T) * hd^-0.5
__global__ __launch_bounds__(256)
void qproj(const float* __restrict__ xq, const float* __restrict__ Wq,
           const float* __restrict__ qnw, const float* __restrict__ qnb,
           float* __restrict__ qvec) {
    __shared__ float xs[DIM];
    const int b = blockIdx.x, t = threadIdx.x;
    xs[t] = xq[(size_t)b * DIM + t];
    __syncthreads();
    const float4* wrow = (const float4*)(Wq + (size_t)t * DIM);
    const float4* xv4  = (const float4*)xs;
    float acc = 0.f;
#pragma unroll 8
    for (int j = 0; j < DIM / 4; ++j) {
        float4 w = wrow[j]; float4 x = xv4[j];
        acc += w.x * x.x + w.y * x.y + w.z * x.z + w.w * x.w;
    }
    float s = acc, s2 = acc * acc;                 // wave32 == one head
#pragma unroll
    for (int m = 16; m >= 1; m >>= 1) {
        s  += __shfl_xor(s,  m, 32);
        s2 += __shfl_xor(s2, m, 32);
    }
    float mu  = s * (1.f / 32.f);
    float var = s2 * (1.f / 32.f) - mu * mu;
    float ri  = rsqrtf(var + EPSF);
    const int lane = t & 31;
    float q = (acc - mu) * ri * qnw[lane] + qnb[lane];
    qvec[(size_t)b * DIM + t] = q * 0.17677669529663687f; // 32^-0.5
}

// ---------------------------------------------------------------- kernel 3
__global__ __launch_bounds__(256, 1)
void attn_fused(const float* __restrict__ xk, const float* __restrict__ xv,
                const float* __restrict__ knw_g, const float* __restrict__ knb_g,
                const float* __restrict__ nw_g, const float* __restrict__ nb_g,
                const float* __restrict__ Wp,
                const _Float16* __restrict__ Wk16,
                const _Float16* __restrict__ Wv16,
                const float* __restrict__ qvec, float* __restrict__ out) {
    __shared__ float    lds_raw[2][16 * DIM];     // async-staged fp32 tiles (32 KB)
    __shared__ _Float16 lds_x[16 * DIM];          // converted f16 tile (8 KB)
    __shared__ float    lds_sc[NHEADS * KLEN];    // scores (16 KB)
    __shared__ float    lds_q[DIM];
    __shared__ float    lds_o[DIM];
    __shared__ float    lds_y[DIM];
    __shared__ float    red[16];

    const int b    = blockIdx.x;
    const int t    = threadIdx.x;
    const int wv   = t >> 5;        // wave id == head id
    const int lane = t & 31;
    const int nlo  = lane & 15;
    const int hi   = lane >> 4;

    lds_q[t] = qvec[(size_t)b * DIM + t];
    __syncthreads();

    const float qh0 = lds_q[wv * HD + nlo];
    const float qh1 = lds_q[wv * HD + 16 + nlo];
    const float kw0 = knw_g[nlo], kw1 = knw_g[16 + nlo];
    const float kb0 = knb_g[nlo], kb1 = knb_g[16 + nlo];

    const int srow = t >> 4;
    const int scol = (t & 15) * 16;

    // this head's Wk slice (256x32) as WMMA B operands: 128 VGPRs
    v16h bw[16];
#pragma unroll
    for (int ks = 0; ks < 8; ++ks)
#pragma unroll
        for (int nh = 0; nh < 2; ++nh) {
            int n = wv * HD + nh * 16 + nlo;
            bw[ks * 2 + nh] = make_op16(Wk16 + (size_t)n * DIM + ks * 32 + hi * 8);
        }

    const float* xk_b = xk + (size_t)b * KLEN * DIM;
    const float* xv_b = xv + (size_t)b * KLEN * DIM;

    // ================= phase 1: K-proj + head-LN + scores =================
    stage_raw(xk_b, lds_raw[0], t);                       // prologue: tile 0
    for (int tile = 0; tile < NTILES; ++tile) {
        const int cur = tile & 1;
        if (tile + 1 < NTILES) {
            stage_raw(xk_b + (size_t)(tile + 1) * 16 * DIM, lds_raw[cur ^ 1], t);
            wait_async4();                                // current tile landed
        } else {
            wait_async0();
        }
        __syncthreads();                                  // raw[cur] visible; prior lds_x reads done

        // convert fp32 tile -> f16 tile (LDS -> LDS)
        {
            const float4* rs = (const float4*)(lds_raw[cur] + srow * DIM + scol);
            float4 f0 = rs[0], f1 = rs[1], f2 = rs[2], f3 = rs[3];
            _Float16* d = lds_x + srow * DIM + scol;
            d[0]  = (_Float16)f0.x; d[1]  = (_Float16)f0.y; d[2]  = (_Float16)f0.z; d[3]  = (_Float16)f0.w;
            d[4]  = (_Float16)f1.x; d[5]  = (_Float16)f1.y; d[6]  = (_Float16)f1.z; d[7]  = (_Float16)f1.w;
            d[8]  = (_Float16)f2.x; d[9]  = (_Float16)f2.y; d[10] = (_Float16)f2.z; d[11] = (_Float16)f2.w;
            d[12] = (_Float16)f3.x; d[13] = (_Float16)f3.y; d[14] = (_Float16)f3.z; d[15] = (_Float16)f3.w;
        }
        __syncthreads();                                  // lds_x ready

        v8f c0 = {}; v8f c1 = {};
#pragma unroll
        for (int ks = 0; ks < 8; ++ks) {
            v16h a = make_op16(lds_x + nlo * DIM + ks * 32 + hi * 8);
            c0 = wmma16(a, bw[ks * 2 + 0], c0);
            c1 = wmma16(a, bw[ks * 2 + 1], c1);
        }

        // per-key-row LayerNorm over 32 head dims, then dot with q_h
        float sc[8];
#pragma unroll
        for (int r = 0; r < 8; ++r) {
            float x0 = c0[r], x1 = c1[r];                 // row M = r + 8*hi
            float s = x0 + x1, s2 = x0 * x0 + x1 * x1;
#pragma unroll
            for (int m = 8; m >= 1; m >>= 1) {
                s  += __shfl_xor(s,  m, 32);
                s2 += __shfl_xor(s2, m, 32);
            }
            float mu  = s * (1.f / 32.f);
            float var = s2 * (1.f / 32.f) - mu * mu;
            float ri  = rsqrtf(var + EPSF);
            float k0  = (x0 - mu) * ri * kw0 + kb0;
            float k1  = (x1 - mu) * ri * kw1 + kb1;
            float d   = k0 * qh0 + k1 * qh1;
#pragma unroll
            for (int m = 8; m >= 1; m >>= 1) d += __shfl_xor(d, m, 32);
            sc[r] = d;
        }
        if (nlo == 0) {
#pragma unroll
            for (int r = 0; r < 8; ++r)
                lds_sc[wv * KLEN + tile * 16 + hi * 8 + r] = sc[r];
        }
    }

    // ================= softmax (wave-local head row of 512) ===============
    __syncthreads();
    float inv_l;
    {
        float m = -1e30f;
        for (int i = lane; i < KLEN; i += 32)
            m = fmaxf(m, lds_sc[wv * KLEN + i]);
#pragma unroll
        for (int s = 16; s >= 1; s >>= 1) m = fmaxf(m, __shfl_xor(m, s, 32));
        float lsum = 0.f;
        for (int i = lane; i < KLEN; i += 32) {
            float e = __expf(lds_sc[wv * KLEN + i] - m);
            lds_sc[wv * KLEN + i] = e;
            lsum += e;
        }
#pragma unroll
        for (int s = 16; s >= 1; s >>= 1) lsum += __shfl_xor(lsum, s, 32);
        inv_l = 1.f / lsum;
    }

    // swap weight registers to this head's Wv slice
#pragma unroll
    for (int ks = 0; ks < 8; ++ks)
#pragma unroll
        for (int nh = 0; nh < 2; ++nh) {
            int n = wv * HD + nh * 16 + nlo;
            bw[ks * 2 + nh] = make_op16(Wv16 + (size_t)n * DIM + ks * 32 + hi * 8);
        }

    // ================= phase 2: V-proj fused with attn @ V ================
    float oA = 0.f, oB = 0.f;
    stage_raw(xv_b, lds_raw[0], t);
    for (int tile = 0; tile < NTILES; ++tile) {
        const int cur = tile & 1;
        if (tile + 1 < NTILES) {
            stage_raw(xv_b + (size_t)(tile + 1) * 16 * DIM, lds_raw[cur ^ 1], t);
            wait_async4();
        } else {
            wait_async0();
        }
        __syncthreads();

        {
            const float4* rs = (const float4*)(lds_raw[cur] + srow * DIM + scol);
            float4 f0 = rs[0], f1 = rs[1], f2 = rs[2], f3 = rs[3];
            _Float16* d = lds_x + srow * DIM + scol;
            d[0]  = (_Float16)f0.x; d[1]  = (_Float16)f0.y; d[2]  = (_Float16)f0.z; d[3]  = (_Float16)f0.w;
            d[4]  = (_Float16)f1.x; d[5]  = (_Float16)f1.y; d[6]  = (_Float16)f1.z; d[7]  = (_Float16)f1.w;
            d[8]  = (_Float16)f2.x; d[9]  = (_Float16)f2.y; d[10] = (_Float16)f2.z; d[11] = (_Float16)f2.w;
            d[12] = (_Float16)f3.x; d[13] = (_Float16)f3.y; d[14] = (_Float16)f3.z; d[15] = (_Float16)f3.w;
        }
        __syncthreads();

        v8f c0 = {}; v8f c1 = {};
#pragma unroll
        for (int ks = 0; ks < 8; ++ks) {
            v16h a = make_op16(lds_x + nlo * DIM + ks * 32 + hi * 8);
            c0 = wmma16(a, bw[ks * 2 + 0], c0);
            c1 = wmma16(a, bw[ks * 2 + 1], c1);
        }
#pragma unroll
        for (int r = 0; r < 8; ++r) {                     // row M = r + 8*hi
            float p = lds_sc[wv * KLEN + tile * 16 + hi * 8 + r];
            oA += p * c0[r];                              // col N = nlo
            oB += p * c1[r];                              // col N = 16+nlo
        }
    }
    oA += __shfl_xor(oA, 16, 32);                         // combine M halves
    oB += __shfl_xor(oB, 16, 32);
    oA *= inv_l; oB *= inv_l;
    if (hi == 0) {
        lds_o[wv * HD + nlo]      = oA;
        lds_o[wv * HD + 16 + nlo] = oB;
    }
    __syncthreads();

    // ================= final LayerNorm(256) + out @ Wp^T ==================
    float v = lds_o[t];
    float s = v, s2 = v * v;
#pragma unroll
    for (int m = 16; m >= 1; m >>= 1) {
        s  += __shfl_xor(s,  m, 32);
        s2 += __shfl_xor(s2, m, 32);
    }
    if (lane == 0) { red[wv] = s; red[8 + wv] = s2; }
    __syncthreads();
    float ts = 0.f, ts2 = 0.f;
#pragma unroll
    for (int i = 0; i < 8; ++i) { ts += red[i]; ts2 += red[8 + i]; }
    float mu  = ts * (1.f / 256.f);
    float var = ts2 * (1.f / 256.f) - mu * mu;
    float ri  = rsqrtf(var + EPSF);
    lds_y[t] = (v - mu) * ri * nw_g[t] + nb_g[t];
    __syncthreads();

    const float4* wrow = (const float4*)(Wp + (size_t)t * DIM);
    const float4* yv4  = (const float4*)lds_y;
    float acc = 0.f;
#pragma unroll 8
    for (int j = 0; j < DIM / 4; ++j) {
        float4 w = wrow[j]; float4 y = yv4[j];
        acc += w.x * y.x + w.y * y.y + w.z * y.z + w.w * y.w;
    }
    out[(size_t)b * DIM + t] = acc;
}

// ---------------------------------------------------------------- launcher
extern "C" void kernel_launch(void* const* d_in, const int* in_sizes, int n_in,
                              void* d_out, int out_size, void* d_ws, size_t ws_size,
                              hipStream_t stream) {
    const float* x_q  = (const float*)d_in[0];
    const float* x_k  = (const float*)d_in[1];
    const float* x_v  = (const float*)d_in[2];
    const float* Wq   = (const float*)d_in[3];
    const float* Wk   = (const float*)d_in[4];
    const float* Wv   = (const float*)d_in[5];
    const float* Wp   = (const float*)d_in[6];
    const float* qn_w = (const float*)d_in[7];
    const float* qn_b = (const float*)d_in[8];
    const float* kn_w = (const float*)d_in[9];
    const float* kn_b = (const float*)d_in[10];
    const float* n_w  = (const float*)d_in[11];
    const float* n_b  = (const float*)d_in[12];

    const int B = 1024;

    // workspace: Wk16 (128KB) | Wv16 (128KB) | qvec (1MB)
    _Float16* wk16 = (_Float16*)d_ws;
    _Float16* wv16 = wk16 + DIM * DIM;
    float*    qvec = (float*)(wv16 + DIM * DIM);

    cvt_weights<<<(DIM * DIM / 4 + 255) / 256, 256, 0, stream>>>(Wk, Wv, wk16, wv16);
    qproj<<<B, 256, 0, stream>>>(x_q, Wq, qn_w, qn_b, qvec);
    attn_fused<<<B, 256, 0, stream>>>(x_k, x_v, kn_w, kn_b, n_w, n_b, Wp,
                                      wk16, wv16, qvec, (float*)d_out);
}
